// Seq2Seq_26044681683100
// MI455X (gfx1250) — matrix operand order repaired and checked by reading, not compiled, not run on hardware
//
#include <hip/hip_runtime.h>
#include <hip/hip_bf16.h>

typedef __attribute__((ext_vector_type(2)))  float  v2f;
typedef __attribute__((ext_vector_type(8)))  float  v8f;
typedef __attribute__((ext_vector_type(16))) __bf16 v16bf;

#define EDIM 256
#define HDIM 512
#define G4H  2048
#define NB   32
#define VOUT 32000
#define HSTR 516   // LDS row stride (floats): 516%64==4 -> conflict-free A-frag reads

// ---------------------------------------------------------------------------
// Pack Whh [2048,512] f32 into WMMA B-fragment order:
//   idx = ((ct*64 + kg)*32 + lane)*4 floats,
//   lane=(lh,hi): floats { k=kg*8+2hi, +1, kg*8+4+2hi, +1 } of column ct*16+lh.
// One b128 per lane then feeds two 16x16x4 WMMAs with zero address math.
// ---------------------------------------------------------------------------
__global__ __launch_bounds__(256) void pack_whh_kernel(
    const float* __restrict__ W, float* __restrict__ P)
{
    const int i = blockIdx.x * 256 + threadIdx.x;      // 128*64*32 = 262144
    const int lane = i & 31, kg = (i >> 5) & 63, ct = i >> 11;
    const int lh = lane & 15, hi = lane >> 4;
    const float* src = W + (long)(ct * 16 + lh) * HDIM + kg * 8 + 2 * hi;
    float4 v;
    v.x = src[0]; v.y = src[1]; v.z = src[4]; v.w = src[5];
    *reinterpret_cast<float4*>(P + (long)i * 4) = v;
}

// ---------------------------------------------------------------------------
// Pack fc_W [32000,512] f32 -> bf16 in WMMA B-fragment order:
//   idx = ((ntg*16 + ks)*32 + lane)*16 bf16,
//   lane=(lh,hi): 16 bf16 { k = ks*32 + 16*hi .. +15 } of column ntg*16+lh.
// ---------------------------------------------------------------------------
__global__ __launch_bounds__(256) void pack_fcw_kernel(
    const float* __restrict__ W, __bf16* __restrict__ P)
{
    const long i = (long)blockIdx.x * 256 + threadIdx.x;   // 2000*16*32 = 1024000
    if (i >= (long)2000 * 16 * 32) return;
    const int lane = (int)(i & 31), ks = (int)((i >> 5) & 15);
    const long nt = i >> 9;
    const int lh = lane & 15, hi = lane >> 4;
    const float* src = W + (nt * 16 + lh) * HDIM + ks * 32 + 16 * hi;
    __bf16* dst = P + i * 16;
#pragma unroll
    for (int j = 0; j < 16; ++j) dst[j] = (__bf16)src[j];
}

// ---------------------------------------------------------------------------
// xg = emb[tok] @ Wih^T + (bih + bhh).  fp32 WMMA 16x16x4 (exact fp32 math).
// ---------------------------------------------------------------------------
__global__ __launch_bounds__(256) void xg_kernel(
    const int* __restrict__ tokens, const float* __restrict__ emb,
    const float* __restrict__ Wih, const float* __restrict__ bih,
    const float* __restrict__ bhh, float* __restrict__ xg,
    int M, int rowT, int srcStride)
{
    const int w = threadIdx.x >> 5, lane = threadIdx.x & 31;
    const int lh = lane & 15, hi = lane >> 4;
    const int flat = blockIdx.x * 8 + w;
    const int mtiles = M >> 4;
    if (flat >= mtiles * (G4H / 16)) return;
    const int mtile = flat / (G4H / 16);
    const int ntile = flat % (G4H / 16);
    const int m0 = mtile * 16, n0 = ntile * 16;

    const int mrow = m0 + lh;
    const int tok  = tokens[(mrow / rowT) * srcStride + (mrow % rowT)];
    const float* arow = emb + (long)tok * EDIM + 2 * hi;
    const float* brow = Wih + (long)(n0 + lh) * EDIM + 2 * hi;

    const float bias = bih[n0 + lh] + bhh[n0 + lh];
    v8f c;
#pragma unroll
    for (int i = 0; i < 8; ++i) c[i] = bias;

#pragma unroll 8
    for (int k0 = 0; k0 < EDIM; k0 += 4) {
        v2f a = *reinterpret_cast<const v2f*>(arow + k0);
        v2f b = *reinterpret_cast<const v2f*>(brow + k0);
        c = __builtin_amdgcn_wmma_f32_16x16x4_f32(false, a, false, b,
                                                  (short)0, c, false, false);
    }
    float* orow = xg + (long)(m0 + 8 * hi) * G4H + n0 + lh;
#pragma unroll
    for (int r = 0; r < 8; ++r) orow[(long)r * G4H] = c[r];
}

// ---------------------------------------------------------------------------
// LSTM recurrence. 2 blocks x 16 batch rows, 16 waves; wave w owns h-cols
// [32w, 32w+32).  c in registers (WMMA D layout), h in LDS, Whh pre-packed:
// per k-group one coalesced b128 feeds two fp32 WMMAs per gate tile.
// ---------------------------------------------------------------------------
__global__ __launch_bounds__(512) void lstm_kernel(
    const float* __restrict__ xg, const float* __restrict__ whp,
    const float* __restrict__ h_in, const float* __restrict__ c_in,
    float* __restrict__ h_out, float* __restrict__ c_out,
    __bf16* __restrict__ hs_bf16, int T, int init_zero, int store_hs)
{
    __shared__ float hbuf[16 * HSTR];
    const int tid = threadIdx.x;
    const int w = tid >> 5, lane = tid & 31, lh = lane & 15, hi = lane >> 4;
    const int bb = blockIdx.x * 16;
    const int j0 = w * 32;

    for (int i = tid; i < 16 * HDIM; i += 512) {
        int m = i >> 9, k = i & (HDIM - 1);
        hbuf[m * HSTR + k] = init_zero ? 0.0f : h_in[(bb + m) * HDIM + k];
    }
    float creg[2][8];
#pragma unroll
    for (int s = 0; s < 2; ++s)
#pragma unroll
        for (int r = 0; r < 8; ++r)
            creg[s][r] = init_zero ? 0.0f
                : c_in[(long)(bb + r + 8 * hi) * HDIM + j0 + s * 16 + lh];
    __syncthreads();

    // packed-B base per gate tile: column tile ct = g*32 + w*2 + s
    const float* bbase[8];
#pragma unroll
    for (int g = 0; g < 4; ++g)
#pragma unroll
        for (int s = 0; s < 2; ++s)
            bbase[g * 2 + s] = whp + (long)(g * 32 + w * 2 + s) * 8192 + lane * 4;

    for (int t = 0; t < T; ++t) {
        v8f acc[8];
#pragma unroll
        for (int g = 0; g < 4; ++g)
#pragma unroll
            for (int s = 0; s < 2; ++s) {
                const int col = g * HDIM + j0 + s * 16 + lh;
                v8f a;
#pragma unroll
                for (int r = 0; r < 8; ++r)
                    a[r] = xg[((long)(bb + r + 8 * hi) * T + t) * G4H + col];
                acc[g * 2 + s] = a;
            }
        for (int kg = 0; kg < 64; ++kg) {
            v2f a0 = *reinterpret_cast<const v2f*>(&hbuf[lh * HSTR + kg * 8 + 2 * hi]);
            v2f a1 = *reinterpret_cast<const v2f*>(&hbuf[lh * HSTR + kg * 8 + 4 + 2 * hi]);
#pragma unroll
            for (int tl = 0; tl < 8; ++tl) {
                union { float4 f; v2f h[2]; } b;
                b.f = *reinterpret_cast<const float4*>(bbase[tl] + (long)kg * 128);
                acc[tl] = __builtin_amdgcn_wmma_f32_16x16x4_f32(
                    false, a0, false, b.h[0], (short)0, acc[tl], false, false);
                acc[tl] = __builtin_amdgcn_wmma_f32_16x16x4_f32(
                    false, a1, false, b.h[1], (short)0, acc[tl], false, false);
            }
        }
        __syncthreads();   // all waves done reading hbuf for step t
#pragma unroll
        for (int s = 0; s < 2; ++s)
#pragma unroll
            for (int r = 0; r < 8; ++r) {
                float iv = acc[0 * 2 + s][r], fv = acc[1 * 2 + s][r];
                float gv = acc[2 * 2 + s][r], ov = acc[3 * 2 + s][r];
                iv = 1.0f / (1.0f + __expf(-iv));
                fv = 1.0f / (1.0f + __expf(-fv));
                gv = tanhf(gv);
                ov = 1.0f / (1.0f + __expf(-ov));
                const float cv = fv * creg[s][r] + iv * gv;
                creg[s][r] = cv;
                const float hv = ov * tanhf(cv);
                hbuf[(r + 8 * hi) * HSTR + j0 + s * 16 + lh] = hv;
                if (store_hs)
                    hs_bf16[((long)(bb + r + 8 * hi) * T + t) * HDIM + j0 + s * 16 + lh]
                        = (__bf16)hv;
            }
        __syncthreads();   // new h visible before next step's reads
    }
    for (int i = tid; i < 16 * HDIM; i += 512) {
        int m = i >> 9, k = i & (HDIM - 1);
        h_out[(bb + m) * HDIM + k] = hbuf[m * HSTR + k];
    }
#pragma unroll
    for (int s = 0; s < 2; ++s)
#pragma unroll
        for (int r = 0; r < 8; ++r)
            c_out[(long)(bb + r + 8 * hi) * HDIM + j0 + s * 16 + lh] = creg[s][r];
}

// ---------------------------------------------------------------------------
// logits = hs @ fc_W^T + fc_b  (M=2528, N=32000, K=512), bf16 WMMA, f32 acc.
// Wave holds its full A strip (16x512 bf16 = 128 VGPRs) and streams 125
// N-tiles of fragment-packed B (fully coalesced), prefetching the next tile.
// ---------------------------------------------------------------------------
__global__ __launch_bounds__(256) void proj_kernel(
    const __bf16* __restrict__ hsb, const __bf16* __restrict__ fwbp,
    const float* __restrict__ fcb, float* __restrict__ out)
{
    const int MT = 158, TDEC = 79;
    const int w = threadIdx.x >> 5, lane = threadIdx.x & 31;
    const int lh = lane & 15, hi = lane >> 4;
    const int mtile = blockIdx.x * 8 + w;
    if (mtile >= MT) return;
    const int m0 = mtile * 16;
    const int nt0 = blockIdx.y * 125;

    // A-frags (bf16 16x32): lane<16 k=[0..15]; lane>=16 k=[16..31]
    v16bf afrag[16];
    {
        const __bf16* arow = hsb + (long)(m0 + lh) * HDIM;
#pragma unroll
        for (int ks = 0; ks < 16; ++ks) {
            const int k0 = ks * 32;
            union { v16bf v; float4 f[2]; } u;
            u.f[0] = *reinterpret_cast<const float4*>(arow + k0 + 16 * hi);
            u.f[1] = *reinterpret_cast<const float4*>(arow + k0 + 16 * hi + 8);
            afrag[ks] = u.v;
        }
    }
    long orow[8];
#pragma unroll
    for (int r = 0; r < 8; ++r) {
        const int rm = m0 + r + 8 * hi;
        const int b = rm / TDEC, t = rm % TDEC;
        orow[r] = (long)(b * 80 + t + 1) * VOUT;
    }
    for (int nt = 0; nt < 125; ++nt) {
        const long ntg = nt0 + nt;
        const __bf16* bfrag = fwbp + ntg * 8192 + (long)lane * 16;
        // speculative prefetch of next tile's packed-B region
        __builtin_prefetch(fwbp + (ntg + 1) * 8192 + (long)lane * 256, 0, 0);
        const int n0 = (int)ntg * 16;
        const float bias = fcb[n0 + lh];
        v8f c;
#pragma unroll
        for (int i = 0; i < 8; ++i) c[i] = bias;
#pragma unroll
        for (int ks = 0; ks < 16; ++ks) {
            union { v16bf v; float4 f[2]; } u;
            u.f[0] = *reinterpret_cast<const float4*>(bfrag + (long)ks * 512);
            u.f[1] = *reinterpret_cast<const float4*>(bfrag + (long)ks * 512 + 8);
            c = __builtin_amdgcn_wmma_f32_16x16x32_bf16(false, afrag[ks], false, u.v,
                                                        (short)0, c, false, false);
        }
#pragma unroll
        for (int r = 0; r < 8; ++r)
            out[orow[r] + n0 + lh] = c[r];
    }
}

__global__ void zero_t0_kernel(float* __restrict__ out)
{
    const int i = blockIdx.x * blockDim.x + threadIdx.x;
    if (i < NB * VOUT) {
        const int b = i / VOUT, v = i % VOUT;
        out[(long)(b * 80) * VOUT + v] = 0.0f;
    }
}

extern "C" void kernel_launch(void* const* d_in, const int* in_sizes, int n_in,
                              void* d_out, int out_size, void* d_ws, size_t ws_size,
                              hipStream_t stream)
{
    const int*   src     = (const int*)d_in[0];
    const int*   trg     = (const int*)d_in[1];
    const float* enc_emb = (const float*)d_in[2];
    const float* enc_Wih = (const float*)d_in[3];
    const float* enc_Whh = (const float*)d_in[4];
    const float* enc_bih = (const float*)d_in[5];
    const float* enc_bhh = (const float*)d_in[6];
    const float* dec_emb = (const float*)d_in[7];
    const float* dec_Wih = (const float*)d_in[8];
    const float* dec_Whh = (const float*)d_in[9];
    const float* dec_bih = (const float*)d_in[10];
    const float* dec_bhh = (const float*)d_in[11];
    const float* fc_W    = (const float*)d_in[12];
    const float* fc_b    = (const float*)d_in[13];
    float* out = (float*)d_out;

    char* ws = (char*)d_ws;
    size_t off = 0;
    float*  xg   = (float*)(ws + off);  off += (size_t)2560 * G4H * 4;   // shared enc/dec
    float*  hA   = (float*)(ws + off);  off += (size_t)NB * HDIM * 4;
    float*  cA   = (float*)(ws + off);  off += (size_t)NB * HDIM * 4;
    float*  hB   = (float*)(ws + off);  off += (size_t)NB * HDIM * 4;
    float*  cB   = (float*)(ws + off);  off += (size_t)NB * HDIM * 4;
    __bf16* hsb  = (__bf16*)(ws + off); off += (size_t)2528 * HDIM * 2;
    __bf16* fwbp = (__bf16*)(ws + off); off += (size_t)VOUT * HDIM * 2;
    float*  whpE = (float*)(ws + off);  off += (size_t)G4H * HDIM * 4;
    float*  whpD = (float*)(ws + off);  off += (size_t)G4H * HDIM * 4;

    // one-time repacks (fragment order)
    pack_fcw_kernel<<<(2000 * 16 * 32 + 255) / 256, 256, 0, stream>>>(fc_W, fwbp);
    pack_whh_kernel<<<(128 * 64 * 32) / 256, 256, 0, stream>>>(enc_Whh, whpE);
    pack_whh_kernel<<<(128 * 64 * 32) / 256, 256, 0, stream>>>(dec_Whh, whpD);

    // encoder: xg then recurrence (h0=c0=0)
    xg_kernel<<<(160 * 128) / 8, 256, 0, stream>>>(src, enc_emb, enc_Wih, enc_bih,
                                                   enc_bhh, xg, 2560, 80, 80);
    lstm_kernel<<<2, 512, 0, stream>>>(xg, whpE, hA, cA, hA, cA, hsb, 80, 1, 0);

    // decoder: xg (reuses buffer) then recurrence seeded by encoder state
    xg_kernel<<<(158 * 128) / 8, 256, 0, stream>>>(trg, dec_emb, dec_Wih, dec_bih,
                                                   dec_bhh, xg, 2528, 79, 80);
    lstm_kernel<<<2, 512, 0, stream>>>(xg, whpD, hA, cA, hB, cB, hsb, 79, 0, 1);

    // output: zero t=0 slice, then bf16-WMMA projection for t=1..79
    zero_t0_kernel<<<(NB * VOUT + 255) / 256, 256, 0, stream>>>(out);
    dim3 g(20, 16);
    proj_kernel<<<g, 256, 0, stream>>>(hsb, fwbp, fc_b, out);
}
